// Decoder_51599737094227
// MI455X (gfx1250) — compile-verified, matrix operand order checked by reading, not run on hardware
//
#include <hip/hip_runtime.h>
#include <hip/hip_bf16.h>

// ---------------- problem constants ----------------
#define B_    32
#define T_    256
#define FEAT_ 640
#define H_    2048
#define P_    640
#define KGATE 1280            // FEAT + P
#define NWG   64              // persistent workgroups (all co-resident)
#define HSLICE 32             // H_ / NWG hidden units per WG
#define KT_GATE 40            // 1280 / 32 k-tiles for gate GEMM
#define NT_GATE 512           // 8192 / 16 n-tiles total
#define KT_PROJ 64            // 2048 / 32
#define NT_PROJ 40            // 640 / 16

typedef __attribute__((ext_vector_type(16))) __bf16 v16bf;
typedef __attribute__((ext_vector_type(8)))  float  v8f;
typedef unsigned int  u32;
typedef unsigned short u16;
typedef __attribute__((ext_vector_type(4))) u32 u32x4;
typedef __attribute__((ext_vector_type(8))) u32 u32x8;

struct bfpair { u32x4 lo; u32x4 hi; };

__device__ __forceinline__ u16 f2bf(float f) {
  u32 u = __builtin_bit_cast(u32, f);
  u += 0x7FFFu + ((u >> 16) & 1u);           // round-to-nearest-even
  return (u16)(u >> 16);
}

__device__ __forceinline__ v16bf frag_cast(u32x4 lo, u32x4 hi) {
  bfpair s; s.lo = lo; s.hi = hi;
  return __builtin_bit_cast(v16bf, s);
}

// fast transcendentals on the TRANS pipe (v_exp_f32), co-executes with WMMA
__device__ __forceinline__ float sigmoidf_(float x) {
  return 1.0f / (1.0f + __expf(-x));
}
__device__ __forceinline__ float tanhf_(float x) {
  // 1 - 2/(e^{2x}+1): exact limits at +/-inf, no inf/inf NaN
  return 1.0f - 2.0f / (__expf(2.0f * x) + 1.0f);
}

// ------------- device-wide barrier (persistent grid of NWG blocks) -------------
__device__ __forceinline__ void grid_sync(u32* bars, u32 k) {
  __syncthreads();
  if (threadIdx.x == 0) {
    __threadfence();
    u32 old = atomicAdd(&bars[0], 1u);
    if (old == k * NWG - 1u) {
      __threadfence();
      __hip_atomic_store(&bars[1], k, __ATOMIC_RELEASE, __HIP_MEMORY_SCOPE_AGENT);
    } else {
      while (__hip_atomic_load(&bars[1], __ATOMIC_ACQUIRE, __HIP_MEMORY_SCOPE_AGENT) < k) {
        __builtin_amdgcn_s_sleep(2);
      }
    }
    __threadfence();
  }
  __syncthreads();
}

// ------------- weight pre-pack: fp32 [K,N] -> bf16 WMMA B-fragment tiles -------------
// Layout: Wp[(nt*ntiles_k + kt)*512 + lane*16 + e]
//   lane: N = nt*16 + (lane&15); klo = lane<16 ? 0 : 8
//   e<8 : K = kt*32 + klo + e ;  e>=8 : K = kt*32 + klo + 16 + (e-8)
__global__ void pack_b_kernel(const float* __restrict__ W, u16* __restrict__ Wp,
                              int Kdim, int Ndim) {
  const int ntk = Kdim >> 5;
  const long total = (long)(Ndim >> 4) * ntk * 512;
  for (long idx = (long)blockIdx.x * blockDim.x + threadIdx.x; idx < total;
       idx += (long)gridDim.x * blockDim.x) {
    int e    = (int)(idx & 15);
    int lane = (int)((idx >> 4) & 31);
    long tile = idx >> 9;
    int kt = (int)(tile % ntk);
    int nt = (int)(tile / ntk);
    int klo = (lane < 16) ? 0 : 8;
    int kk  = (e < 8) ? (klo + e) : (klo + 8 + e);
    int k = kt * 32 + kk;
    int n = nt * 16 + (lane & 15);
    Wp[idx] = f2bf(W[(long)k * Ndim + n]);
  }
}

// ------------- elementwise fp32 -> bf16 (input tensor) -------------
__global__ void pack_x_kernel(const float* __restrict__ x, u16* __restrict__ xb, long n) {
  for (long i = (long)blockIdx.x * blockDim.x + threadIdx.x; i < n;
       i += (long)gridDim.x * blockDim.x) {
    xb[i] = f2bf(x[i]);
  }
}

// ------------- per-layer init: zero barrier + recurrent m state -------------
__global__ void init_kernel(u32* __restrict__ bars, u16* __restrict__ m_buf) {
  int tid = blockIdx.x * blockDim.x + threadIdx.x;
  if (tid < 8) bars[tid] = 0u;
  for (int i = tid; i < B_ * P_; i += blockDim.x * gridDim.x) m_buf[i] = 0;
}

// ------------- persistent LSTMP layer kernel -------------
__launch_bounds__(256, 1)
__global__ void lstmp_layer_kernel(
    const u16* __restrict__ in,       // [B,T,640] bf16 input sequence
    const u16* __restrict__ Wp,       // packed gate weights  [512 nt][40 kt][32][16]
    const float* __restrict__ bias,   // [8192]
    const u16* __restrict__ Pp,       // packed proj weights  [40 nt][64 kt][32][16]
    u16* __restrict__ y_bf16,         // [B,T,640] bf16 out (layer 0) or null
    float* __restrict__ y_f32,        // [B,T,640] fp32 out (layer 1) or null
    u16* __restrict__ m_buf,          // [B,640] bf16 recurrent projection state
    u16* __restrict__ hact,           // [B,2048] bf16 gated hidden activations
    u32* __restrict__ bars)
{
  __shared__ u16   lds_a[B_ * 640];        // 40 KB: activation half-panel (bf16)
  __shared__ float lds_s[4096];            // 16 KB: gate z's / proj partials
  __shared__ float lds_c[B_ * HSLICE];     //  4 KB: persistent cell state slice

  const int tid  = threadIdx.x;
  const int wave = tid >> 5;
  const int lane = tid & 31;
  const int wg   = blockIdx.x;

  for (int i = tid; i < B_ * HSLICE; i += 256) lds_c[i] = 0.0f;

  // gate-GEMM role: wave -> one 16-wide column tile of one gate
  const int gate = wave >> 1;                               // 0..3 = i,j,f,o
  const int sub  = wave & 1;
  const int ncol0   = gate * H_ + wg * HSLICE + sub * 16;   // first of 16 cols
  const int ntile_g = ncol0 >> 4;
  const float bcol  = bias[ncol0 + (lane & 15)];

  const int klo  = (lane < 16) ? 0 : 8;
  const int arow = lane & 15;

  for (int t = 0; t < T_; ++t) {
    // ---------------- Phase A: z = [x_t | m_{t-1}] @ W + b ----------------
    v8f acc0, acc1;
    #pragma unroll
    for (int r = 0; r < 8; ++r) { acc0[r] = bcol; acc1[r] = bcol; }

    for (int half = 0; half < 2; ++half) {
      __syncthreads();
      // stage bf16 half-panel [32][640] via 16-byte vector copies (10 iter/thread)
      if (half == 0) {
        #pragma unroll
        for (int it = 0; it < 10; ++it) {
          int c   = tid + it * 256;           // chunk id, 2560 total
          int row = c / 80;                   // batch row
          int c8  = c - row * 80;             // 16B chunk within row
          u32x4 v = *(const u32x4*)(in + ((long)row * T_ + t) * 640 + c8 * 8);
          *(u32x4*)&lds_a[c * 8] = v;
        }
      } else {
        #pragma unroll
        for (int it = 0; it < 10; ++it) {
          int c = tid + it * 256;
          u32x4 v = *(const u32x4*)(m_buf + c * 8);
          *(u32x4*)&lds_a[c * 8] = v;
        }
      }
      __syncthreads();

      const u16* wp = Wp + (size_t)(ntile_g * KT_GATE + half * 20) * 512 + lane * 16;
      #pragma unroll 4
      for (int kt = 0; kt < 20; ++kt) {
        u32x8 bw = *(const u32x8*)wp;  wp += 512;            // B frag (global/L2)
        v16bf bf = __builtin_bit_cast(v16bf, bw);
        int kb = kt * 32 + klo;
        const u16* a0p = &lds_a[arow * 640 + kb];
        const u16* a1p = &lds_a[(arow + 16) * 640 + kb];
        v16bf af0 = frag_cast(*(const u32x4*)a0p, *(const u32x4*)(a0p + 16));
        v16bf af1 = frag_cast(*(const u32x4*)a1p, *(const u32x4*)(a1p + 16));
        acc0 = __builtin_amdgcn_wmma_f32_16x16x32_bf16(false, af0, false, bf,
                                                       (short)0, acc0, false, false);
        acc1 = __builtin_amdgcn_wmma_f32_16x16x32_bf16(false, af1, false, bf,
                                                       (short)0, acc1, false, false);
      }
    }

    // ---------------- Phase B: gate nonlinearities + cell update ----------------
    __syncthreads();
    {
      const int colh = sub * 16 + (lane & 15);
      const int badd = (lane < 16) ? 0 : 8;
      float* zg = &lds_s[gate * 1024];
      #pragma unroll
      for (int r = 0; r < 8; ++r) {
        zg[(r + badd) * 32 + colh]        = acc0[r];
        zg[(16 + r + badd) * 32 + colh]   = acc1[r];
      }
    }
    __syncthreads();

    #pragma unroll
    for (int it = 0; it < 4; ++it) {
      int e = tid + it * 256;                 // 1024 (batch, h) elements
      int b = e >> 5, h = e & 31;
      float zi = lds_s[0 * 1024 + b * 32 + h];
      float zj = lds_s[1 * 1024 + b * 32 + h];
      float zf = lds_s[2 * 1024 + b * 32 + h];
      float zo = lds_s[3 * 1024 + b * 32 + h];
      float c_new = sigmoidf_(zf + 1.0f) * lds_c[e] + sigmoidf_(zi) * tanhf_(zj);
      lds_c[e] = c_new;
      float hval = sigmoidf_(zo) * tanhf_(c_new);
      hact[b * H_ + wg * HSLICE + h] = f2bf(hval);
    }

    grid_sync(bars, 2u * t + 1u);

    // ---------------- Phase C: m = h @ P (WGs 0..39) ----------------
    if (wg < NT_PROJ) {
      const int nt = wg;
      v8f p0, p1;
      #pragma unroll
      for (int r = 0; r < 8; ++r) { p0[r] = 0.0f; p1[r] = 0.0f; }

      const u16* pp = Pp + (size_t)(nt * KT_PROJ + wave * 8) * 512 + lane * 16;
      #pragma unroll
      for (int kt = 0; kt < 8; ++kt) {
        int kg = wave * 8 + kt;
        u32x8 bw = *(const u32x8*)pp;  pp += 512;
        v16bf bf = __builtin_bit_cast(v16bf, bw);
        int kb = kg * 32 + klo;
        const u16* h0 = &hact[arow * H_ + kb];
        const u16* h1 = &hact[(arow + 16) * H_ + kb];
        v16bf af0 = frag_cast(*(const u32x4*)h0, *(const u32x4*)(h0 + 16));
        v16bf af1 = frag_cast(*(const u32x4*)h1, *(const u32x4*)(h1 + 16));
        p0 = __builtin_amdgcn_wmma_f32_16x16x32_bf16(false, af0, false, bf,
                                                     (short)0, p0, false, false);
        p1 = __builtin_amdgcn_wmma_f32_16x16x32_bf16(false, af1, false, bf,
                                                     (short)0, p1, false, false);
      }
      {
        const int n = lane & 15;
        const int badd = (lane < 16) ? 0 : 8;
        float* dst = &lds_s[wave * 512];
        #pragma unroll
        for (int r = 0; r < 8; ++r) {
          dst[(r + badd) * 16 + n]      = p0[r];
          dst[(16 + r + badd) * 16 + n] = p1[r];
        }
      }
      __syncthreads();
      #pragma unroll
      for (int it = 0; it < 2; ++it) {
        int e = tid + it * 256;                // 512 output elements
        float s = 0.0f;
        #pragma unroll
        for (int w = 0; w < 8; ++w) s += lds_s[w * 512 + e];
        int b = e >> 4, n = nt * 16 + (e & 15);
        m_buf[b * 640 + n] = f2bf(s);
        if (y_bf16) y_bf16[((long)b * T_ + t) * 640 + n] = f2bf(s);
        if (y_f32)  y_f32 [((long)b * T_ + t) * 640 + n] = s;
      }
    }

    grid_sync(bars, 2u * t + 2u);
  }
}

// ---------------- host-side launch ----------------
extern "C" void kernel_launch(void* const* d_in, const int* in_sizes, int n_in,
                              void* d_out, int out_size, void* d_ws, size_t ws_size,
                              hipStream_t stream) {
  (void)in_sizes; (void)n_in; (void)out_size; (void)ws_size;
  const float* x  = (const float*)d_in[0];
  const float* W0 = (const float*)d_in[1];
  const float* b0 = (const float*)d_in[2];
  const float* P0 = (const float*)d_in[3];
  const float* W1 = (const float*)d_in[4];
  const float* b1 = (const float*)d_in[5];
  const float* P1 = (const float*)d_in[6];
  float* out = (float*)d_out;

  char* ws = (char*)d_ws;
  size_t off = 0;
  auto carve = [&](size_t bytes) -> char* {
    char* p = ws + off;
    off = (off + bytes + 255) & ~(size_t)255;
    return p;
  };
  const size_t W_PACK = (size_t)NT_GATE * KT_GATE * 512 * sizeof(u16); // 20.97 MB
  const size_t P_PACK = (size_t)NT_PROJ * KT_PROJ * 512 * sizeof(u16); //  2.62 MB
  u16* W0p  = (u16*)carve(W_PACK);
  u16* W1p  = (u16*)carve(W_PACK);
  u16* P0p  = (u16*)carve(P_PACK);
  u16* P1p  = (u16*)carve(P_PACK);
  u16* xb   = (u16*)carve((size_t)B_ * T_ * FEAT_ * sizeof(u16));      // 10.5 MB
  u16* y0   = (u16*)carve((size_t)B_ * T_ * P_ * sizeof(u16));         // 10.5 MB
  u16* mbuf = (u16*)carve((size_t)B_ * P_ * sizeof(u16));
  u16* hact = (u16*)carve((size_t)B_ * H_ * sizeof(u16));
  u32* bars = (u32*)carve(256);

  // 1) pack weights + input fp32 -> bf16 (weights into WMMA fragment layout;
  //    bf16 working set ~47 MB stays resident in the 192 MB L2)
  pack_b_kernel<<<2048, 256, 0, stream>>>(W0, W0p, KGATE, 4 * H_);
  pack_b_kernel<<<2048, 256, 0, stream>>>(W1, W1p, KGATE, 4 * H_);
  pack_b_kernel<<<512,  256, 0, stream>>>(P0, P0p, H_, P_);
  pack_b_kernel<<<512,  256, 0, stream>>>(P1, P1p, H_, P_);
  pack_x_kernel<<<2048, 256, 0, stream>>>(x, xb, (long)B_ * T_ * FEAT_);

  // 2) layer 0 (bf16 input -> bf16 intermediate)
  init_kernel<<<80, 256, 0, stream>>>(bars, mbuf);
  lstmp_layer_kernel<<<NWG, 256, 0, stream>>>(
      xb, W0p, b0, P0p,
      /*y_bf16=*/y0, /*y_f32=*/nullptr, mbuf, hact, bars);

  // 3) layer 1 (bf16 intermediate -> fp32 output)
  init_kernel<<<80, 256, 0, stream>>>(bars, mbuf);
  lstmp_layer_kernel<<<NWG, 256, 0, stream>>>(
      y0, W1p, b1, P1p,
      /*y_bf16=*/nullptr, /*y_f32=*/out, mbuf, hact, bars);
}
